// mha_77962246357464
// MI455X (gfx1250) — matrix operand order repaired and checked by reading, not compiled, not run on hardware
//
#include <hip/hip_runtime.h>
#include <hip/hip_bf16.h>

// ---------------------------------------------------------------------------
// Fused: qkv = x@Wqkv^T + b ; RoPE(q,k) ; 4-head causal attention (HD=32) ;
//        y@Wfc2^T + b ; SiLU.   bf16 WMMA (v_wmma_f32_16x16x32_bf16) for all
//        GEMMs, f32 accumulation, softmax in f32 registers.
// One workgroup (8 wave32) per batch element, everything staged in LDS.
// ---------------------------------------------------------------------------

typedef __bf16 bf16_t;
typedef __attribute__((ext_vector_type(16))) __bf16 v16bf;
typedef __attribute__((ext_vector_type(8)))  float  v8f;
typedef __attribute__((ext_vector_type(4)))  unsigned int v4u;

#define LSEQ   119
#define DMODEL 128
#define NHEADS 4
#define HDIM   32
#define XS     136   // bf16 row stride for sX/sAttn/sVT/sY  (272B: 16B-aligned, bank offset 4)
#define QS     40    // bf16 row stride for sQ/sK            (80B : 16B-aligned, bank offset 20)

// LDS layout (bytes)
#define OFF_X   0                          // [128][136] bf16  (reused as attn)
#define OFF_Q   34816                      // [4][128][40] bf16
#define OFF_K   (34816 + 40960)            // [4][128][40] bf16
#define OFF_VT  (34816 + 81920)            // [4][32][136] bf16 (transposed v)
#define OFF_Y   (34816 + 81920 + 34816)    // [128][136] bf16
#define LDS_BYTES (34816 + 81920 + 34816 + 34816)   // 186368

// Native converts (gfx1250 has full bf16 VALU; fptrunc/fpext lower to v_cvt).
__device__ __forceinline__ bf16_t f2bf(float f) { return (bf16_t)f; }
__device__ __forceinline__ float  bf2f(bf16_t b) { return (float)b; }

// A-matrix 16x32 bf16 fragment (ISA 7.12.2): lanes 0-15 row=lane, K kb..kb+7 & kb+16..kb+23
// with kb = 0 (lanes 0-15) / 8 (lanes 16-31). Source is row-major [rows][stride].
__device__ __forceinline__ v16bf load_frag_A(const bf16_t* base, int row0, int stride,
                                             int k0, int lane) {
  int m  = lane & 15;
  int kb = (lane < 16) ? 0 : 8;
  const bf16_t* p = base + (row0 + m) * stride + k0 + kb;
  union { v4u u[2]; v16bf v; } c;
  c.u[0] = *(const v4u*)(p);        // K = kb .. kb+7
  c.u[1] = *(const v4u*)(p + 16);   // K = kb+16 .. kb+23
  return c.v;
}

// B-matrix 32x16 bf16 fragment: lanes 0-15 col=lane hold K 0..7 & 16..23,
// lanes 16-31 hold K 8..15 & 24..31. Source is "B-transposed" row-major [N][stride] over K.
__device__ __forceinline__ v16bf load_frag_B(const bf16_t* base, int col0, int stride,
                                             int k0, int lane) {
  int n    = lane & 15;
  int koff = (lane < 16) ? 0 : 8;
  const bf16_t* p = base + (col0 + n) * stride + k0 + koff;
  union { v4u u[2]; v16bf v; } c;
  c.u[0] = *(const v4u*)(p);        // K = koff .. koff+7
  c.u[1] = *(const v4u*)(p + 16);   // K = koff+16 .. koff+23
  return c.v;
}

__device__ __forceinline__ v8f wmma_bf16(v16bf a, v16bf b, v8f c) {
  return __builtin_amdgcn_wmma_f32_16x16x32_bf16(false, a, false, b, (short)0, c,
                                                 false, false);
}

// ---------------------------------------------------------------------------
// Prep: weights -> bf16 in workspace; RoPE tables (bf16-rounded, stored f32).
// ---------------------------------------------------------------------------
__global__ void prep_kernel(const float* __restrict__ Wqkv, const float* __restrict__ Wfc2,
                            bf16_t* __restrict__ wq, bf16_t* __restrict__ wf,
                            float* __restrict__ cosT, float* __restrict__ sinT) {
  int i = blockIdx.x * blockDim.x + threadIdx.x;
  if (i < 3 * DMODEL * DMODEL) wq[i] = f2bf(Wqkv[i]);
  if (i < DMODEL * DMODEL)     wf[i] = f2bf(Wfc2[i]);
  if (i < LSEQ * 64) {
    int t = i >> 6, f = i & 63;
    // inv_freq = 10000^(-(2f)/128) = exp(-(2f/128) * ln(10000))
    float inv = __expf(-((float)(2 * f) / 128.0f) * 9.210340371976184f);
    float ang = (float)t * inv;
    cosT[i] = bf2f(f2bf(cosf(ang)));
    sinT[i] = bf2f(f2bf(sinf(ang)));
  }
}

// ---------------------------------------------------------------------------
// Main fused kernel: one block per batch element.
// ---------------------------------------------------------------------------
__global__ __launch_bounds__(256)
void fused_mha_kernel(const float* __restrict__ x,
                      const float* __restrict__ bqkv,
                      const float* __restrict__ bfc2,
                      const bf16_t* __restrict__ Wq,
                      const bf16_t* __restrict__ Wf,
                      const float* __restrict__ cosT,
                      const float* __restrict__ sinT,
                      float* __restrict__ out) {
  extern __shared__ char smem[];
  bf16_t* sX    = (bf16_t*)(smem + OFF_X);    // x (bf16), later reused as attn
  bf16_t* sQ    = (bf16_t*)(smem + OFF_Q);
  bf16_t* sK    = (bf16_t*)(smem + OFF_K);
  bf16_t* sVT   = (bf16_t*)(smem + OFF_VT);
  bf16_t* sY    = (bf16_t*)(smem + OFF_Y);
  bf16_t* sAttn = sX;

  const int tid  = threadIdx.x;
  const int wv   = tid >> 5;
  const int lane = tid & 31;
  const int nbase = lane & 15;
  const int b    = blockIdx.x;

  // ---- Phase 0: stage x (bf16), zero-pad rows 119..127 -------------------
  const float* xb = x + (size_t)b * LSEQ * DMODEL;
  for (int idx = tid; idx < 128 * 128; idx += 256) {
    int r = idx >> 7, c = idx & 127;
    float v = (r < LSEQ) ? xb[r * DMODEL + c] : 0.0f;
    sX[r * XS + c] = f2bf(v);
  }
  __syncthreads();

  // ---- Phase 1: qkv = x @ Wqkv^T + b  (M=128, N=384, K=128) --------------
  for (int nt = wv; nt < 24; nt += 8) {
    int ng   = nt * 16 + nbase;                 // output feature 0..383
    float bias = bqkv[ng];
    v16bf Bf[4];
    #pragma unroll
    for (int kk = 0; kk < 4; ++kk)
      Bf[kk] = load_frag_B(Wq, nt * 16, DMODEL, kk * 32, lane);

    int which = ng >> 7;                        // 0=q 1=k 2=v (uniform per tile)
    int c = ng & 127;
    int h = c & 3, d = c >> 2;                  // head-strided split: c = 4d + h

    for (int mt = 0; mt < 8; ++mt) {
      v8f acc = {0.f, 0.f, 0.f, 0.f, 0.f, 0.f, 0.f, 0.f};
      #pragma unroll
      for (int kk = 0; kk < 4; ++kk) {
        v16bf Af = load_frag_A(sX, mt * 16, XS, kk * 32, lane);
        acc = wmma_bf16(Af, Bf[kk], acc);
      }
      int mb = mt * 16 + ((lane < 16) ? 0 : 8);
      if (which == 2) {                         // v -> transposed, packed b32x4
        union { unsigned u[4]; bf16_t h2[8]; } pk;
        #pragma unroll
        for (int i = 0; i < 8; ++i) pk.h2[i] = f2bf(acc[i] + bias);
        unsigned* dst = (unsigned*)(sVT + (h * HDIM + d) * XS + mb);  // mb even
        #pragma unroll
        for (int i = 0; i < 4; ++i) dst[i] = pk.u[i];
      } else {                                  // q/k -> head-major (pre-RoPE)
        bf16_t* dst = ((which == 0) ? sQ : sK) + (h * 128) * QS + d;
        #pragma unroll
        for (int i = 0; i < 8; ++i) dst[(mb + i) * QS] = f2bf(acc[i] + bias);
      }
    }
  }
  __syncthreads();

  // ---- Phase 1b: RoPE on q,k (pairs d, d+16 within a head) ---------------
  for (int idx = tid; idx < 2 * NHEADS * 16 * LSEQ; idx += 256) {
    int t = idx;
    int d = t & 15;  t >>= 4;
    int h = t & 3;   t >>= 2;
    int pos = t % LSEQ;
    int qk  = t / LSEQ;
    bf16_t* M  = (qk == 0) ? sQ : sK;
    bf16_t* p1 = M + (h * 128 + pos) * QS + d;
    bf16_t* p2 = p1 + 16;
    float x1 = bf2f(*p1), x2 = bf2f(*p2);
    int fi = pos * 64 + 4 * d + h;              // freq index = low-half channel
    float co = cosT[fi], si = sinT[fi];
    *p1 = f2bf(x1 * co + x2 * si);
    *p2 = f2bf(x2 * co - x1 * si);
  }
  __syncthreads();

  // ---- Phase 2: per-head causal attention --------------------------------
  const float scale = 0.17677669529663687f;     // 1/sqrt(32)
  const int mt = wv;                            // this wave's 16 query rows
  const int mb = mt * 16 + ((lane < 16) ? 0 : 8);

  for (int h = 0; h < NHEADS; ++h) {
    const bf16_t* Qh = sQ + (h * 128) * QS;
    const bf16_t* Kh = sK + (h * 128) * QS;
    const bf16_t* Vh = sVT + (h * HDIM) * XS;

    // scores: one 16x16x32 WMMA per N-tile (K = HDIM = 32)
    v16bf Aq = load_frag_A(Qh, mt * 16, QS, 0, lane);
    v8f s[8];
    #pragma unroll
    for (int ntg = 0; ntg < 8; ++ntg) {
      v16bf Bk = load_frag_B(Kh, ntg * 16, QS, 0, lane);
      v8f a = {0.f, 0.f, 0.f, 0.f, 0.f, 0.f, 0.f, 0.f};
      s[ntg] = wmma_bf16(Aq, Bk, a);
    }
    // scale + causal/valid mask (coordinates per lane, no divergence)
    #pragma unroll
    for (int ntg = 0; ntg < 8; ++ntg) {
      int n = ntg * 16 + nbase;
      #pragma unroll
      for (int i = 0; i < 8; ++i) {
        int m = mb + i;
        float v = s[ntg][i] * scale;
        s[ntg][i] = (n <= m && n < LSEQ) ? v : -__builtin_inff();
      }
    }
    // softmax across this wave's rows: reduce over 8 reg-groups + 16 lanes
    #pragma unroll
    for (int i = 0; i < 8; ++i) {
      float mx = s[0][i];
      #pragma unroll
      for (int ntg = 1; ntg < 8; ++ntg) mx = fmaxf(mx, s[ntg][i]);
      #pragma unroll
      for (int off = 1; off < 16; off <<= 1) mx = fmaxf(mx, __shfl_xor(mx, off, 32));
      float sum = 0.f;
      #pragma unroll
      for (int ntg = 0; ntg < 8; ++ntg) {
        float p = __expf(s[ntg][i] - mx);
        s[ntg][i] = p;
        sum += p;
      }
      #pragma unroll
      for (int off = 1; off < 16; off <<= 1) sum += __shfl_xor(sum, off, 32);
      float r = 1.0f / sum;
      #pragma unroll
      for (int ntg = 0; ntg < 8; ++ntg) s[ntg][i] *= r;
    }
    // attn -> LDS (bf16) for A-fragment re-layout (own rows only)
    #pragma unroll
    for (int ntg = 0; ntg < 8; ++ntg) {
      int n = ntg * 16 + nbase;
      #pragma unroll
      for (int i = 0; i < 8; ++i) sAttn[(mb + i) * XS + n] = f2bf(s[ntg][i]);
    }
    __syncthreads();

    // y_h = attn @ v   (M=128, N=32, K=128), v pre-transposed in sVT
    v8f y0 = {0.f, 0.f, 0.f, 0.f, 0.f, 0.f, 0.f, 0.f};
    v8f y1 = {0.f, 0.f, 0.f, 0.f, 0.f, 0.f, 0.f, 0.f};
    #pragma unroll
    for (int kk = 0; kk < 4; ++kk) {
      v16bf Aa = load_frag_A(sAttn, mt * 16, XS, kk * 32, lane);
      v16bf B0 = load_frag_B(Vh, 0,  XS, kk * 32, lane);
      v16bf B1 = load_frag_B(Vh, 16, XS, kk * 32, lane);
      y0 = wmma_bf16(Aa, B0, y0);
      y1 = wmma_bf16(Aa, B1, y1);
    }
    // scatter back into model channel order c = 4d + h
    #pragma unroll
    for (int i = 0; i < 8; ++i) {
      int m = mb + i;
      sY[m * XS + 4 * nbase + h]        = f2bf(y0[i]);
      sY[m * XS + 4 * (16 + nbase) + h] = f2bf(y1[i]);
    }
    __syncthreads();
  }

  // ---- Phase 3: out = silu(Y @ Wfc2^T + b) -------------------------------
  float* ob = out + (size_t)b * LSEQ * DMODEL;
  for (int nt = 0; nt < 8; ++nt) {
    int ng = nt * 16 + nbase;
    float bias = bfc2[ng];
    v8f acc = {0.f, 0.f, 0.f, 0.f, 0.f, 0.f, 0.f, 0.f};
    #pragma unroll
    for (int kk = 0; kk < 4; ++kk) {
      v16bf Ay = load_frag_A(sY, mt * 16, XS, kk * 32, lane);
      v16bf Bw = load_frag_B(Wf, nt * 16, DMODEL, kk * 32, lane);
      acc = wmma_bf16(Ay, Bw, acc);
    }
    #pragma unroll
    for (int i = 0; i < 8; ++i) {
      int m = mb + i;
      if (m < LSEQ) {
        float v = acc[i] + bias;
        ob[m * DMODEL + ng] = v / (1.0f + __expf(-v));   // silu
      }
    }
  }
}

// ---------------------------------------------------------------------------
extern "C" void kernel_launch(void* const* d_in, const int* in_sizes, int n_in,
                              void* d_out, int out_size, void* d_ws, size_t ws_size,
                              hipStream_t stream) {
  const float* x    = (const float*)d_in[0];
  const float* Wqkv = (const float*)d_in[1];
  const float* bqkv = (const float*)d_in[2];
  const float* Wfc2 = (const float*)d_in[3];
  const float* bfc2 = (const float*)d_in[4];
  float* out = (float*)d_out;

  char* ws = (char*)d_ws;
  bf16_t* wq   = (bf16_t*)(ws);                               // 98304 B
  bf16_t* wf   = (bf16_t*)(ws + 98304);                       // 32768 B
  float*  cosT = (float*)(ws + 98304 + 32768);                // 30464 B
  float*  sinT = (float*)(ws + 98304 + 32768 + 30464);        // 30464 B

  prep_kernel<<<192, 256, 0, stream>>>(Wqkv, Wfc2, wq, wf, cosT, sinT);

  static_assert(LDS_BYTES == 186368, "lds layout");
  hipFuncSetAttribute((const void*)fused_mha_kernel,
                      hipFuncAttributeMaxDynamicSharedMemorySize, LDS_BYTES);
  fused_mha_kernel<<<2048, 256, LDS_BYTES, stream>>>(x, bqkv, bfc2, wq, wf,
                                                     cosT, sinT, out);
}
